// GroupQueryAttention_11768210391732
// MI455X (gfx1250) — compile-verified, matrix operand order checked by reading, not run on hardware
//
#include <hip/hip_runtime.h>

#define D_MODEL 1024
#define NUM_HEADS 16
#define NUM_GROUPS 4
#define D_K 64
#define BATCH 4
#define SEQ 1024
#define KV_DIM (NUM_GROUPS * D_K)   // 256
#define ROWS (BATCH * SEQ)          // 4096

typedef __attribute__((ext_vector_type(16))) _Float16 v16h;
typedef __attribute__((ext_vector_type(8)))  _Float16 v8h;
typedef __attribute__((ext_vector_type(8)))  float    v8f;

// ---- fragment loaders (wave32 WMMA 16-bit A/B layout) -----------------------
// Lane L: halves 0..7 = K (hi8)+0..7, halves 8..15 = K (hi8)+16..23, hi8 = 8*(L>=16).
// Caller passes p = row_base + k_base + hi8 (multiple of 8 halves -> 16B aligned).
__device__ __forceinline__ v16h load_frag_f16(const _Float16* p) {
    v8h lo = *(const v8h*)p;
    v8h hi = *(const v8h*)(p + 16);
    v16h r;
#pragma unroll
    for (int i = 0; i < 8; ++i) { r[i] = lo[i]; r[i + 8] = hi[i]; }
    return r;
}

__device__ __forceinline__ v16h load_frag_f32(const float* p) {
    v16h r;
#pragma unroll
    for (int i = 0; i < 8; ++i) {
        r[i]     = (_Float16)p[i];
        r[i + 8] = (_Float16)p[i + 16];
    }
    return r;
}

// ---- elementwise f32 -> f16 -------------------------------------------------
__global__ void cvt_f32_f16_kernel(const float* __restrict__ in,
                                   _Float16* __restrict__ out, int n) {
    int i = blockIdx.x * blockDim.x + threadIdx.x;
    if (i < n) out[i] = (_Float16)in[i];
}

// ---- w[in,out] f32 -> wT[out,in] f16 ---------------------------------------
__global__ void transpose_cvt_kernel(const float* __restrict__ w,
                                     _Float16* __restrict__ wt,
                                     int in_dim, int out_dim) {
    int i = blockIdx.x * blockDim.x + threadIdx.x;
    if (i >= in_dim * out_dim) return;
    int o = i % out_dim;
    int r = i / out_dim;
    wt[(size_t)o * in_dim + r] = (_Float16)w[i];
}

// ---- generic WMMA GEMM: Y = X[rows,K] @ WT[N,K]^T + bias --------------------
// mode 0: Y f16 row-major [rows,N]
// mode 1: Y f32 row-major [rows,N]
// mode 2: Y f16 transposed per batch: Y[b][n][s], b = row/SEQ, s = row%SEQ
__global__ void gemm_bias_kernel(const _Float16* __restrict__ X,
                                 const _Float16* __restrict__ WT,
                                 const float* __restrict__ bias,
                                 void* __restrict__ Y,
                                 int rows, int K, int N, int mode) {
    const int lane = threadIdx.x;                       // 0..31
    const int wid  = blockIdx.x * blockDim.y + threadIdx.y;
    const int ntiles = N >> 4;
    const int total  = (rows >> 4) * ntiles;
    if (wid >= total) return;
    const int tn   = wid % ntiles;
    const int tm   = wid / ntiles;
    const int mrow = (tm << 4) + (lane & 15);
    const int ncol = (tn << 4) + (lane & 15);
    const int hi8  = (lane >> 4) << 3;

    const _Float16* pa = X  + (size_t)mrow * K + hi8;
    const _Float16* pb = WT + (size_t)ncol * K + hi8;

    v8f acc = {};
    for (int k = 0; k < K; k += 32) {
        v16h a = load_frag_f16(pa + k);
        v16h b = load_frag_f16(pb + k);
        acc = __builtin_amdgcn_wmma_f32_16x16x32_f16(false, a, false, b,
                                                     (short)0, acc, false, false);
    }

    const float bv = bias[ncol];
    const int   m0 = (tm << 4) + hi8;   // C/D: VGPR r -> M = m0 + r, N = ncol
    if (mode == 0) {
        _Float16* out = (_Float16*)Y;
#pragma unroll
        for (int r = 0; r < 8; ++r)
            out[(size_t)(m0 + r) * N + ncol] = (_Float16)(acc[r] + bv);
    } else if (mode == 1) {
        float* out = (float*)Y;
#pragma unroll
        for (int r = 0; r < 8; ++r)
            out[(size_t)(m0 + r) * N + ncol] = acc[r] + bv;
    } else {
        _Float16* out = (_Float16*)Y;
#pragma unroll
        for (int r = 0; r < 8; ++r) {
            int row = m0 + r;
            int b   = row >> 10;           // SEQ = 1024
            int s   = row & (SEQ - 1);
            out[((size_t)b * N + ncol) * SEQ + s] = (_Float16)(acc[r] + bv);
        }
    }
}

// ---- scores = scale * Q @ K^T, written raw into attn region -----------------
__global__ void scores_kernel(const _Float16* __restrict__ qh,  // [B,S,1024] f16
                              const _Float16* __restrict__ kh,  // [B,S,256]  f16
                              float* __restrict__ attn,         // [B*H,S,S]  f32
                              float scale) {
    const int lane = threadIdx.x;
    const int wid  = blockIdx.x * blockDim.y + threadIdx.y;
    const int QT = SEQ >> 4, KT = SEQ >> 4;        // 64, 64
    const int per_bh = QT * KT;
    if (wid >= BATCH * NUM_HEADS * per_bh) return;
    const int bh  = wid / per_bh;
    const int rem = wid % per_bh;
    const int tq = rem / KT, tk = rem % KT;
    const int b = bh >> 4, h = bh & 15, g = h >> 2;
    const int qrow = (tq << 4) + (lane & 15);
    const int krow = (tk << 4) + (lane & 15);
    const int hi8  = (lane >> 4) << 3;

    const _Float16* pa = qh + ((size_t)b * SEQ + qrow) * D_MODEL + h * D_K + hi8;
    const _Float16* pb = kh + ((size_t)b * SEQ + krow) * KV_DIM  + g * D_K + hi8;

    v8f acc = {};
#pragma unroll
    for (int k = 0; k < D_K; k += 32) {
        v16h a  = load_frag_f16(pa + k);
        v16h bf = load_frag_f16(pb + k);
        acc = __builtin_amdgcn_wmma_f32_16x16x32_f16(false, a, false, bf,
                                                     (short)0, acc, false, false);
    }

    float* out = attn + (size_t)bh * SEQ * SEQ;
    const int m0 = (tq << 4) + hi8;
    const int nc = (tk << 4) + (lane & 15);
#pragma unroll
    for (int r = 0; r < 8; ++r)
        out[(size_t)(m0 + r) * SEQ + nc] = acc[r] * scale;
}

// ---- in-place row softmax, one wave per row of 1024 -------------------------
__global__ void softmax_kernel(float* __restrict__ attn, int nrows) {
    const int lane = threadIdx.x & 31;
    const int row  = blockIdx.x * (blockDim.x >> 5) + (threadIdx.x >> 5);
    if (row >= nrows) return;
    float* p = attn + (size_t)row * SEQ;

    float vals[SEQ / 32];
    float m = -3.4e38f;
#pragma unroll
    for (int i = 0; i < SEQ / 32; ++i) {
        vals[i] = p[lane + i * 32];
        m = fmaxf(m, vals[i]);
    }
#pragma unroll
    for (int off = 16; off > 0; off >>= 1) m = fmaxf(m, __shfl_xor(m, off, 32));
    float s = 0.0f;
#pragma unroll
    for (int i = 0; i < SEQ / 32; ++i) { vals[i] = __expf(vals[i] - m); s += vals[i]; }
#pragma unroll
    for (int off = 16; off > 0; off >>= 1) s += __shfl_xor(s, off, 32);
    const float inv = 1.0f / s;
#pragma unroll
    for (int i = 0; i < SEQ / 32; ++i) p[lane + i * 32] = vals[i] * inv;
}

// ---- ctx = attn @ V (A converted f32->f16 in-register, B = V transposed) ----
__global__ void av_kernel(const float* __restrict__ attn,     // [B*H,S,S] f32
                          const _Float16* __restrict__ vt,    // [B,256,S] f16
                          _Float16* __restrict__ ctx) {       // [B,S,1024] f16
    const int lane = threadIdx.x;
    const int wid  = blockIdx.x * blockDim.y + threadIdx.y;
    const int QT = SEQ >> 4;       // 64
    const int NT = D_K >> 4;       // 4
    const int per_bh = QT * NT;    // 256
    if (wid >= BATCH * NUM_HEADS * per_bh) return;
    const int bh  = wid / per_bh;
    const int rem = wid % per_bh;
    const int tq = rem / NT, tn = rem % NT;
    const int b = bh >> 4, h = bh & 15, g = h >> 2;
    const int qrow  = (tq << 4) + (lane & 15);
    const int dkcol = (tn << 4) + (lane & 15);
    const int hi8   = (lane >> 4) << 3;

    const float*    pa = attn + ((size_t)bh * SEQ + qrow) * SEQ + hi8;
    const _Float16* pb = vt + ((size_t)b * KV_DIM + g * D_K + dkcol) * SEQ + hi8;

    v8f acc = {};
    for (int k = 0; k < SEQ; k += 32) {
        v16h a  = load_frag_f32(pa + k);
        v16h bf = load_frag_f16(pb + k);
        acc = __builtin_amdgcn_wmma_f32_16x16x32_f16(false, a, false, bf,
                                                     (short)0, acc, false, false);
    }

    const int m0 = (tq << 4) + hi8;
#pragma unroll
    for (int r = 0; r < 8; ++r) {
        int qv = m0 + r;
        ctx[((size_t)b * SEQ + qv) * D_MODEL + h * D_K + dkcol] = (_Float16)acc[r];
    }
}

extern "C" void kernel_launch(void* const* d_in, const int* in_sizes, int n_in,
                              void* d_out, int out_size, void* d_ws, size_t ws_size,
                              hipStream_t stream) {
    (void)in_sizes; (void)n_in; (void)out_size; (void)ws_size;
    // Inputs (setup_inputs order). NOTE: q (d_in[0]) is unused — the reference
    // faithfully applies the Q projection to the `k` input.
    const float* k_in = (const float*)d_in[1];
    const float* v_in = (const float*)d_in[2];
    const float* w_q  = (const float*)d_in[3];
    const float* b_q  = (const float*)d_in[4];
    const float* w_k  = (const float*)d_in[5];
    const float* b_k  = (const float*)d_in[6];
    const float* w_v  = (const float*)d_in[7];
    const float* b_v  = (const float*)d_in[8];
    const float* w_o  = (const float*)d_in[9];
    const float* b_o  = (const float*)d_in[10];

    float* out_ptr  = (float*)d_out;                                   // [B,S,1024]
    float* attn_ptr = out_ptr + (size_t)BATCH * SEQ * D_MODEL;         // [B,H,S,S]

    // Workspace carve-up (all 256B aligned).
    char*  ws  = (char*)d_ws;
    size_t off = 0;
    auto carve = [&](size_t bytes) -> void* {
        void* p = ws + off;
        off = (off + bytes + 255) & ~(size_t)255;
        return p;
    };
    _Float16* x16 = (_Float16*)carve((size_t)ROWS * D_MODEL * 2);   // k input, f16
    _Float16* y16 = (_Float16*)carve((size_t)ROWS * D_MODEL * 2);   // v input, f16
    _Float16* wqT = (_Float16*)carve((size_t)D_MODEL * D_MODEL * 2);
    _Float16* wkT = (_Float16*)carve((size_t)KV_DIM * D_MODEL * 2);
    _Float16* wvT = (_Float16*)carve((size_t)KV_DIM * D_MODEL * 2);
    _Float16* woT = (_Float16*)carve((size_t)D_MODEL * D_MODEL * 2);
    _Float16* qh  = (_Float16*)carve((size_t)ROWS * D_MODEL * 2);   // [B,S,1024]
    _Float16* kh  = (_Float16*)carve((size_t)ROWS * KV_DIM * 2);    // [B,S,256]
    _Float16* vt  = (_Float16*)carve((size_t)BATCH * KV_DIM * SEQ * 2); // [B,256,S]
    _Float16* ctx = (_Float16*)carve((size_t)ROWS * D_MODEL * 2);   // [B,S,1024]

    const dim3 blk1(256);
    const dim3 blkw(32, 8);   // 8 waves per workgroup, one 16x16 tile per wave

    // 1) f32 -> f16 converts of activations
    {
        int n = ROWS * D_MODEL;
        cvt_f32_f16_kernel<<<(n + 255) / 256, blk1, 0, stream>>>(k_in, x16, n);
        cvt_f32_f16_kernel<<<(n + 255) / 256, blk1, 0, stream>>>(v_in, y16, n);
    }
    // 2) weight transposes (w[in,out] f32 -> wT[out,in] f16)
    {
        int n = D_MODEL * D_MODEL;
        transpose_cvt_kernel<<<(n + 255) / 256, blk1, 0, stream>>>(w_q, wqT, D_MODEL, D_MODEL);
        transpose_cvt_kernel<<<(n + 255) / 256, blk1, 0, stream>>>(w_o, woT, D_MODEL, D_MODEL);
        int m = D_MODEL * KV_DIM;
        transpose_cvt_kernel<<<(m + 255) / 256, blk1, 0, stream>>>(w_k, wkT, D_MODEL, KV_DIM);
        transpose_cvt_kernel<<<(m + 255) / 256, blk1, 0, stream>>>(w_v, wvT, D_MODEL, KV_DIM);
    }
    // 3) projections (qh from k input — faithful to reference)
    {
        int tiles = (ROWS / 16) * (D_MODEL / 16);                // 16384
        gemm_bias_kernel<<<tiles / 8, blkw, 0, stream>>>(x16, wqT, b_q, qh,
                                                         ROWS, D_MODEL, D_MODEL, 0);
        int tkv = (ROWS / 16) * (KV_DIM / 16);                   // 4096
        gemm_bias_kernel<<<tkv / 8, blkw, 0, stream>>>(x16, wkT, b_k, kh,
                                                       ROWS, D_MODEL, KV_DIM, 0);
        gemm_bias_kernel<<<tkv / 8, blkw, 0, stream>>>(y16, wvT, b_v, vt,
                                                       ROWS, D_MODEL, KV_DIM, 2);
    }
    // 4) raw scores -> attn region of d_out
    {
        int tiles = BATCH * NUM_HEADS * (SEQ / 16) * (SEQ / 16); // 262144
        scores_kernel<<<tiles / 8, blkw, 0, stream>>>(qh, kh, attn_ptr, 0.125f);
    }
    // 5) softmax in place (one wave per row)
    {
        int nrows = BATCH * NUM_HEADS * SEQ;                     // 65536
        softmax_kernel<<<nrows / 8, blk1, 0, stream>>>(attn_ptr, nrows);
    }
    // 6) ctx = attn @ V
    {
        int tiles = BATCH * NUM_HEADS * (SEQ / 16) * (D_K / 16); // 16384
        av_kernel<<<tiles / 8, blkw, 0, stream>>>(attn_ptr, vt, ctx);
    }
    // 7) out = ctx @ w_o + b_o (f32 straight into d_out)
    {
        int tiles = (ROWS / 16) * (D_MODEL / 16);                // 16384
        gemm_bias_kernel<<<tiles / 8, blkw, 0, stream>>>(ctx, woT, b_o, out_ptr,
                                                         ROWS, D_MODEL, D_MODEL, 1);
    }
}
